// AdditiveAttention_65429531787458
// MI455X (gfx1250) — matrix operand order repaired and checked by reading, not compiled, not run on hardware
//
#include <hip/hip_runtime.h>
#include <hip/hip_bf16.h>
#include <stdint.h>

// ---------------------------------------------------------------------------
// AdditiveAttention on MI455X (gfx1250, wave32, WMMA)
//   score GEMM: X[131072,256] x bf16(w1+w2)[256,256], fused tanh + wv dot
//   v_wmma_f32_16x16x32_bf16, B staged in LDS (batched ds_load clause),
//   branchless tanh epilogue
// ---------------------------------------------------------------------------

typedef __attribute__((ext_vector_type(16))) __bf16 v16bf;
typedef __attribute__((ext_vector_type(8)))  float  v8f;

__device__ __forceinline__ uint32_t pk_bf16(float a, float b) {
    // round-to-nearest-even f32 -> bf16, pack (a -> [15:0], b -> [31:16])
    uint32_t ua = __float_as_uint(a);
    uint32_t ub = __float_as_uint(b);
    ua += 0x7FFFu + ((ua >> 16) & 1u);
    ub += 0x7FFFu + ((ub >> 16) & 1u);
    return (ua >> 16) | (ub & 0xFFFF0000u);
}

// branchless fast tanh: 1 - 2/(e^{2x}+1), via v_exp_f32 + v_rcp_f32
__device__ __forceinline__ float fast_tanh(float x) {
    const float e = __builtin_amdgcn_exp2f(x * 2.8853900817779268f); // e^(2x)
    return 1.0f - 2.0f * __builtin_amdgcn_rcpf(e + 1.0f);
}

// branchless fast exp via v_exp_f32
__device__ __forceinline__ float fast_exp(float x) {
    return __builtin_amdgcn_exp2f(x * 1.4426950408889634f);
}

// ---------------------------------------------------------------------------
// Kernel 1: pack bf16(w1+w2) into WMMA B-matrix tile order.
// Tile (nt,ks) covers N = nt*16..+15, K = ks*32..+31.
// Per ISA 7.12.2 (16-bit B, 32x16): lane<16 -> N=lane, VGPR 0..3 -> K=0..7, VGPR 4..7 -> K=16..23;
// lane>=16 -> N=lane-16, VGPR 0..3 -> K=8..15, VGPR 4..7 -> K=24..31.
// Storage: Wpk[((nt*8+ks)*32 + lane)*8 + v]  (8 contiguous dwords per lane)
// ---------------------------------------------------------------------------
__global__ __launch_bounds__(256)
void pack_weights(const float* __restrict__ w1, const float* __restrict__ b1,
                  const float* __restrict__ w2, const float* __restrict__ b2,
                  uint32_t* __restrict__ Wpk, float* __restrict__ bsum) {
    const int blk = blockIdx.x;          // 0..127 = nt*8 + ks
    const int tid = threadIdx.x;         // 0..255
    const int lane = tid >> 3;           // 0..31
    const int j    = tid & 7;            // 0..7 (VGPR index)
    const int nt = blk >> 3;
    const int ks = blk & 7;
    const int n  = nt * 16 + (lane & 15);
    const int k0 = ks * 32 + ((lane & 16) ? 8 : 0) + ((j < 4) ? 2 * j : 16 + 2 * (j - 4));
    const float x0 = w1[k0 * 256 + n]       + w2[k0 * 256 + n];
    const float x1 = w1[(k0 + 1) * 256 + n] + w2[(k0 + 1) * 256 + n];
    Wpk[((blk * 32) + lane) * 8 + j] = pk_bf16(x0, x1);
    if (blk == 0) bsum[tid] = b1[tid] + b2[tid];
}

// ---------------------------------------------------------------------------
// Kernel 2: fused score GEMM.
//  block = 256 threads = 8 waves; each wave does 16 rows -> 128 rows / block
//  grid  = 131072 / 128 = 1024 blocks
//  B staged to LDS in two 64KB halves (8 N-tiles each). Per N-tile: one
//  clause of 16 ds_load_b128 -> single s_wait_dscnt -> 8 back-to-back WMMAs.
// ---------------------------------------------------------------------------
__global__ __launch_bounds__(256)
void score_kernel(const float* __restrict__ feat, const uint32_t* __restrict__ Wpk,
                  const float* __restrict__ bsum, const float* __restrict__ wv,
                  const float* __restrict__ bvp, float* __restrict__ scoreOut) {
    __shared__ uint4 Blds[4096];         // 64 KB: 8 ntiles x 8 ks x 32 lanes x 2 uint4

    const int tid  = threadIdx.x;
    const int lane = tid & 31;
    const int wave = tid >> 5;
    const int rh   = lane & 15;
    const bool hi  = (lane & 16) != 0;
    const long rBase = (long)blockIdx.x * 128 + (long)wave * 16;
    const float* frow = feat + (rBase + rh) * 256;

    // Preload A tiles (ISA 7.12.2, 16-bit A 16x32):
    //  lane<16: M=lane, VGPR 0..3 -> K=0..7 (pairs), VGPR 4..7 -> K=16..23
    //  lane>=16:         VGPR 0..3 -> K=8..15,        VGPR 4..7 -> K=24..31
    uint32_t Areg[8][8];
#pragma unroll
    for (int ks = 0; ks < 8; ++ks) {
        const int kb = ks * 32 + (hi ? 8 : 0);
#pragma unroll
        for (int j = 0; j < 8; ++j) {
            const int col = kb + ((j < 4) ? 2 * j : 16 + 2 * (j - 4));
            const float2 f = *(const float2*)(frow + col);
            Areg[ks][j] = pk_bf16(f.x, f.y);
        }
    }

    float partial[8];
#pragma unroll
    for (int v = 0; v < 8; ++v) partial[v] = 0.0f;
    const float bv0 = bvp[0];

    for (int half = 0; half < 2; ++half) {
        // stage 8 N-tiles worth of B (64KB) cooperatively into LDS
        const uint4* src = (const uint4*)Wpk + (long)half * 4096;
        for (int i = tid; i < 4096; i += 256) Blds[i] = src[i];
        __syncthreads();

#pragma unroll
        for (int ntl = 0; ntl < 8; ++ntl) {
            const int nt  = half * 8 + ntl;
            const int n   = nt * 16 + rh;
            const float bn  = bsum[n];
            const float wvn = wv[n];

            // Load ALL B tiles for this N-tile first: 16 x ds_load_b128 into
            // distinct registers -> one clause, one dscnt wait, then 8 WMMAs
            // back-to-back (only accumulator chaining between them).
            uint4 Bq[16];
            const uint4* bp = Blds + ((ntl * 8) * 32 + lane) * 2;
#pragma unroll
            for (int ks = 0; ks < 8; ++ks) {
                Bq[2 * ks]     = bp[0];
                Bq[2 * ks + 1] = bp[1];
                bp += 64;                // next K-step tile: 32 lanes * 2 uint4
            }

            v8f acc = {0.f, 0.f, 0.f, 0.f, 0.f, 0.f, 0.f, 0.f};
#pragma unroll
            for (int ks = 0; ks < 8; ++ks) {
                union { uint4 q[2]; v16bf v; } B;
                B.q[0] = Bq[2 * ks];
                B.q[1] = Bq[2 * ks + 1];
                union { uint32_t u[8]; v16bf v; } A;
#pragma unroll
                for (int j = 0; j < 8; ++j) A.u[j] = Areg[ks][j];
                acc = __builtin_amdgcn_wmma_f32_16x16x32_bf16(
                    false, A.v, false, B.v, (short)0, acc, false, false);
            }

            // epilogue: h = tanh(acc + b), partial += h * wv[n]  (branchless)
#pragma unroll
            for (int v = 0; v < 8; ++v) {
                partial[v] += fast_tanh(acc[v] + bn) * wvn;
            }
        }
        __syncthreads();                 // protect LDS before next stage
    }

    // reduce across 16 lanes of each half (xor masks < 16 keep halves separate)
#pragma unroll
    for (int m = 1; m < 16; m <<= 1) {
#pragma unroll
        for (int v = 0; v < 8; ++v)
            partial[v] += __shfl_xor(partial[v], m, 32);
    }
    if (rh == 0) {
        float* out = scoreOut + rBase + (hi ? 8 : 0);
#pragma unroll
        for (int v = 0; v < 8; ++v) out[v] = partial[v] + bv0;
    }
}

// ---------------------------------------------------------------------------
// Kernel 3: softmax over S=4096, in place, per batch. Deterministic.
// ---------------------------------------------------------------------------
__global__ __launch_bounds__(256)
void softmax_kernel(float* __restrict__ attn) {
    const int b = blockIdx.x;
    float* s = attn + (long)b * 4096;
    __shared__ float red[256];
    const int tid = threadIdx.x;

    float mx = -3.402823466e+38f;
    for (int i = tid; i < 4096; i += 256) mx = fmaxf(mx, s[i]);
    red[tid] = mx; __syncthreads();
    for (int st = 128; st > 0; st >>= 1) {
        if (tid < st) red[tid] = fmaxf(red[tid], red[tid + st]);
        __syncthreads();
    }
    mx = red[0]; __syncthreads();

    float sum = 0.0f;
    for (int i = tid; i < 4096; i += 256) sum += fast_exp(s[i] - mx);
    red[tid] = sum; __syncthreads();
    for (int st = 128; st > 0; st >>= 1) {
        if (tid < st) red[tid] += red[tid + st];
        __syncthreads();
    }
    const float inv = __builtin_amdgcn_rcpf(red[0]);
    for (int i = tid; i < 4096; i += 256) s[i] = fast_exp(s[i] - mx) * inv;
}

// ---------------------------------------------------------------------------
// Kernel 4: context partials: partial[b][ch][c] = sum over 256 s of attn * X
// grid (32,16) x 256 threads, coalesced over c.
// ---------------------------------------------------------------------------
__global__ __launch_bounds__(256)
void ctx_partial(const float* __restrict__ feat, const float* __restrict__ attn,
                 float* __restrict__ partial) {
    const int b = blockIdx.x;
    const int ch = blockIdx.y;
    const int c = threadIdx.x;
    const long sBase = (long)b * 4096 + (long)ch * 256;
    const float* f = feat + sBase * 256 + c;
    const float* a = attn + sBase;
    float acc = 0.0f;
    for (int s = 0; s < 256; ++s) acc = fmaf(a[s], f[(long)s * 256], acc);
    partial[(b * 16 + ch) * 256 + c] = acc;
}

__global__ __launch_bounds__(256)
void ctx_final(const float* __restrict__ partial, float* __restrict__ ctx) {
    const int b = blockIdx.x;
    const int c = threadIdx.x;
    float acc = 0.0f;
#pragma unroll
    for (int ch = 0; ch < 16; ++ch) acc += partial[(b * 16 + ch) * 256 + c];
    ctx[b * 256 + c] = acc;
}

// ---------------------------------------------------------------------------
extern "C" void kernel_launch(void* const* d_in, const int* in_sizes, int n_in,
                              void* d_out, int out_size, void* d_ws, size_t ws_size,
                              hipStream_t stream) {
    (void)in_sizes; (void)n_in; (void)out_size; (void)ws_size;
    const float* feat = (const float*)d_in[0];   // [32,4096,256]
    const float* w1   = (const float*)d_in[1];   // [256,256]
    const float* b1   = (const float*)d_in[2];   // [256]
    const float* w2   = (const float*)d_in[3];   // [256,256]
    const float* b2   = (const float*)d_in[4];   // [256]
    const float* wv   = (const float*)d_in[5];   // [256,1]
    const float* bv   = (const float*)d_in[6];   // [1]

    float* ctx  = (float*)d_out;                 // [32,256]
    float* attn = ctx + 32 * 256;                // [32,4096,1]

    uint32_t* Wpk  = (uint32_t*)d_ws;                           // 128 KB
    float* bsum    = (float*)((char*)d_ws + 131072);            // 1 KB
    float* partial = (float*)((char*)d_ws + 131072 + 1024);     // 512 KB

    pack_weights<<<128, 256, 0, stream>>>(w1, b1, w2, b2, Wpk, bsum);
    score_kernel<<<1024, 256, 0, stream>>>(feat, Wpk, bsum, wv, bv, attn);
    softmax_kernel<<<32, 256, 0, stream>>>(attn);
    ctx_partial<<<dim3(32, 16), 256, 0, stream>>>(feat, attn, partial);
    ctx_final<<<32, 256, 0, stream>>>(partial, ctx);
}
